// MultiHeadAttention_7387343749436
// MI455X (gfx1250) — matrix operand order repaired and checked by reading, not compiled, not run on hardware
//
#include <hip/hip_runtime.h>

typedef __attribute__((ext_vector_type(16))) __bf16 v16bf;
typedef __attribute__((ext_vector_type(8)))  float  v8f;

#define S_LEN 2048
#define D_DIM 1024
#define H_NUM 16
#define HD    64
#define B_NUM 4
#define M_ROWS (B_NUM * S_LEN)  // 8192

union AFrag { v16bf v; uint4 q[2]; };

struct FalseT { static constexpr bool value = false; };
struct TrueT  { static constexpr bool value = true;  };

__device__ __forceinline__ unsigned short f2bf(float f) {
  union { float f; unsigned int u; } c; c.f = f;
  unsigned int u = c.u;
  unsigned int r = u + 0x7FFFu + ((u >> 16) & 1u);  // round-to-nearest-even
  return (unsigned short)(r >> 16);
}

__device__ __forceinline__ v8f vzero() {
  v8f z;
#pragma unroll
  for (int i = 0; i < 8; ++i) z[i] = 0.0f;
  return z;
}

__device__ __forceinline__ v8f wmma_bf16(v16bf a, v16bf b, v8f c) {
  return __builtin_amdgcn_wmma_f32_16x16x32_bf16(false, a, false, b, (short)0, c,
                                                 false, false);
}

// ------------------------------------------------- fp32 -> bf16, 8 elems/thread
__global__ void cvt_bf16_kernel(const float* __restrict__ in,
                                unsigned int* __restrict__ out, int n8) {
  int i = blockIdx.x * blockDim.x + threadIdx.x;
  const int stride = gridDim.x * blockDim.x;
  for (; i < n8; i += stride) {
    const float4 f0 = ((const float4*)in)[2 * i];
    const float4 f1 = ((const float4*)in)[2 * i + 1];
    uint4 o;
    o.x = (unsigned int)f2bf(f0.x) | ((unsigned int)f2bf(f0.y) << 16);
    o.y = (unsigned int)f2bf(f0.z) | ((unsigned int)f2bf(f0.w) << 16);
    o.z = (unsigned int)f2bf(f1.x) | ((unsigned int)f2bf(f1.y) << 16);
    o.w = (unsigned int)f2bf(f1.z) | ((unsigned int)f2bf(f1.w) << 16);
    ((uint4*)out)[i] = o;
  }
}

// ------------------------------------------------------------------- GEMM
// Y[m,n] = sum_k A[m,k]*W[n,k] + bias[n]  (bf16 in, f32 acc)
// MODE 0: bf16 row-major; MODE 1: bf16 [B,H,HD,S] (V transposed); MODE 2: f32.
struct GBuf { AFrag a0, a1, b0, b1, b2, b3; };

__device__ __forceinline__ void gload(GBuf& g, const unsigned short* ar0,
                                      const unsigned short* ar1,
                                      const unsigned short* w0,
                                      const unsigned short* w1,
                                      const unsigned short* w2,
                                      const unsigned short* w3, int k) {
  g.a0.q[0] = *(const uint4*)(ar0 + k); g.a0.q[1] = *(const uint4*)(ar0 + k + 8);
  g.a1.q[0] = *(const uint4*)(ar1 + k); g.a1.q[1] = *(const uint4*)(ar1 + k + 8);
  g.b0.q[0] = *(const uint4*)(w0 + k);  g.b0.q[1] = *(const uint4*)(w0 + k + 8);
  g.b1.q[0] = *(const uint4*)(w1 + k);  g.b1.q[1] = *(const uint4*)(w1 + k + 8);
  g.b2.q[0] = *(const uint4*)(w2 + k);  g.b2.q[1] = *(const uint4*)(w2 + k + 8);
  g.b3.q[0] = *(const uint4*)(w3 + k);  g.b3.q[1] = *(const uint4*)(w3 + k + 8);
}

__device__ __forceinline__ void gstep(const GBuf& g, v8f acc[2][4]) {
  acc[0][0] = wmma_bf16(g.a0.v, g.b0.v, acc[0][0]);
  acc[0][1] = wmma_bf16(g.a0.v, g.b1.v, acc[0][1]);
  acc[0][2] = wmma_bf16(g.a0.v, g.b2.v, acc[0][2]);
  acc[0][3] = wmma_bf16(g.a0.v, g.b3.v, acc[0][3]);
  acc[1][0] = wmma_bf16(g.a1.v, g.b0.v, acc[1][0]);
  acc[1][1] = wmma_bf16(g.a1.v, g.b1.v, acc[1][1]);
  acc[1][2] = wmma_bf16(g.a1.v, g.b2.v, acc[1][2]);
  acc[1][3] = wmma_bf16(g.a1.v, g.b3.v, acc[1][3]);
}

template <int MODE>
__global__ void __launch_bounds__(128)
gemm_kernel(const unsigned short* __restrict__ A,
            const unsigned short* __restrict__ W,
            const float* __restrict__ bias, void* __restrict__ outv) {
  const int lane = threadIdx.x & 31;
  const int wave = threadIdx.x >> 5;
  const int hlf  = lane >> 4;
  const int lcol = lane & 15;
  const int m0 = blockIdx.x * 128 + wave * 32;  // two 16-row tiles
  const int n0 = blockIdx.y * 64;

  v8f acc[2][4];
#pragma unroll
  for (int rt = 0; rt < 2; ++rt)
#pragma unroll
    for (int t = 0; t < 4; ++t) acc[rt][t] = vzero();

  const unsigned short* ar0 = A + (size_t)(m0 + lcol) * D_DIM + hlf * 16;
  const unsigned short* ar1 = ar0 + (size_t)16 * D_DIM;
  const unsigned short* w0 = W + (size_t)(n0 + lcol) * D_DIM + hlf * 16;
  const unsigned short* w1 = w0 + (size_t)16 * D_DIM;
  const unsigned short* w2 = w0 + (size_t)32 * D_DIM;
  const unsigned short* w3 = w0 + (size_t)48 * D_DIM;

  GBuf g0, g1;
  gload(g0, ar0, ar1, w0, w1, w2, w3, 0);
  gload(g1, ar0, ar1, w0, w1, w2, w3, 32);
#pragma clang loop unroll(disable)
  for (int k0 = 0; k0 < D_DIM - 64; k0 += 64) {
    gstep(g0, acc);
    gload(g0, ar0, ar1, w0, w1, w2, w3, k0 + 64);
    gstep(g1, acc);
    gload(g1, ar0, ar1, w0, w1, w2, w3, k0 + 96);
  }
  gstep(g0, acc);
  gstep(g1, acc);

#pragma unroll
  for (int rt = 0; rt < 2; ++rt) {
#pragma unroll
    for (int t = 0; t < 4; ++t) {
      const int col = n0 + t * 16 + lcol;
      const float bv = bias[col];
#pragma unroll
      for (int r = 0; r < 8; ++r) {
        const int row = m0 + rt * 16 + r + hlf * 8;
        const float v = acc[rt][t][r] + bv;
        if (MODE == 2) {
          ((float*)outv)[(size_t)row * D_DIM + col] = v;
        } else if (MODE == 1) {
          const int b = row >> 11, s = row & (S_LEN - 1);
          const int h = col >> 6, hd = col & 63;
          ((unsigned short*)outv)[((size_t)(b * H_NUM + h) * HD + hd) * S_LEN + s] =
              f2bf(v);
        } else {
          ((unsigned short*)outv)[(size_t)row * D_DIM + col] = f2bf(v);
        }
      }
    }
  }
}

// ------------------------------------------------------- causal flash attn
__device__ __forceinline__ void load_kfrags(AFrag kf[4],
                                            const unsigned short* Kh, int j0,
                                            int lcol, int hlf) {
#pragma unroll
  for (int nt = 0; nt < 2; ++nt)
#pragma unroll
    for (int ks = 0; ks < 2; ++ks) {
      const unsigned short* kp =
          Kh + (size_t)(j0 + nt * 16 + lcol) * D_DIM + ks * 32 + hlf * 16;
      kf[nt * 2 + ks].q[0] = *(const uint4*)(kp);
      kf[nt * 2 + ks].q[1] = *(const uint4*)(kp + 8);
    }
}

__global__ void __launch_bounds__(128)
attn_kernel(const unsigned short* __restrict__ Qb,
            const unsigned short* __restrict__ Kb,
            const unsigned short* __restrict__ Vt,
            unsigned short* __restrict__ Ao) {
  __shared__ __align__(16) unsigned short Ps[4][16 * 32];
  const int lane = threadIdx.x & 31;
  const int wave = threadIdx.x >> 5;
  const int hlf  = lane >> 4;
  const int lcol = lane & 15;
  const int bh = blockIdx.y;
  const int b = bh >> 4, h = bh & 15;
  const int q0 = blockIdx.x * 64 + wave * 16;

  const unsigned short* Qh = Qb + (size_t)b * S_LEN * D_DIM + h * HD;
  const unsigned short* Kh = Kb + (size_t)b * S_LEN * D_DIM + h * HD;
  const unsigned short* Vh = Vt + (size_t)bh * HD * S_LEN;
  unsigned short* Pw = Ps[wave];

  // Q A-fragments for the two K=32 steps of the HD=64 contraction
  AFrag qa[2];
  {
    const unsigned short* qp = Qh + (size_t)(q0 + lcol) * D_DIM + hlf * 16;
    qa[0].q[0] = *(const uint4*)(qp);
    qa[0].q[1] = *(const uint4*)(qp + 8);
    qa[1].q[0] = *(const uint4*)(qp + 32);
    qa[1].q[1] = *(const uint4*)(qp + 40);
  }

  v8f o[4];
#pragma unroll
  for (int t = 0; t < 4; ++t) o[t] = vzero();
  float mr[8], lr[8];
#pragma unroll
  for (int r = 0; r < 8; ++r) { mr[r] = -3.0e38f; lr[r] = 0.0f; }

  const float sc = 0.125f * 1.44269504f;  // 1/sqrt(HD) * log2(e)

  AFrag kf[4];
  load_kfrags(kf, Kh, 0, lcol, hlf);

  // Process one 32-key block. MASKED=true only for the single diagonal block.
  auto body = [&](auto mc, int j0) {
    constexpr bool MASKED = decltype(mc)::value;

    // scores: Q(16x64) @ K^T(64x32)
    v8f s0 = vzero(), s1 = vzero();
    s0 = wmma_bf16(qa[0].v, kf[0].v, s0);
    s0 = wmma_bf16(qa[1].v, kf[1].v, s0);
    s1 = wmma_bf16(qa[0].v, kf[2].v, s1);
    s1 = wmma_bf16(qa[1].v, kf[3].v, s1);

    // issue V loads (this block) and, for non-final blocks, K loads for the
    // next block; the softmax VALU section hides their latency
    AFrag vf[4];
#pragma unroll
    for (int t = 0; t < 4; ++t) {
      const unsigned short* vp =
          Vh + (size_t)(t * 16 + lcol) * S_LEN + j0 + hlf * 16;
      vf[t].q[0] = *(const uint4*)(vp);
      vf[t].q[1] = *(const uint4*)(vp + 8);
    }
    if (!MASKED) load_kfrags(kf, Kh, j0 + 32, lcol, hlf);

    // (optional causal mask) + online softmax in base-2
    float p0[8], p1[8], alpha[8];
#pragma unroll
    for (int r = 0; r < 8; ++r) {
      float ss0 = s0[r] * sc;
      float ss1 = s1[r] * sc;
      if (MASKED) {
        const int rowg = q0 + r + hlf * 8;
        if (j0 + lcol > rowg)      ss0 = -3.0e38f;
        if (j0 + 16 + lcol > rowg) ss1 = -3.0e38f;
      }
      float bm = fmaxf(ss0, ss1);
      bm = fmaxf(bm, __shfl_xor(bm, 1, 32));
      bm = fmaxf(bm, __shfl_xor(bm, 2, 32));
      bm = fmaxf(bm, __shfl_xor(bm, 4, 32));
      bm = fmaxf(bm, __shfl_xor(bm, 8, 32));
      const float nm = fmaxf(mr[r], bm);
      alpha[r] = exp2f(mr[r] - nm);
      mr[r] = nm;
      const float e0 = exp2f(ss0 - nm);
      const float e1 = exp2f(ss1 - nm);
      float rs = e0 + e1;
      rs += __shfl_xor(rs, 1, 32);
      rs += __shfl_xor(rs, 2, 32);
      rs += __shfl_xor(rs, 4, 32);
      rs += __shfl_xor(rs, 8, 32);
      lr[r] = lr[r] * alpha[r] + rs;
      p0[r] = e0; p1[r] = e1;
    }

    // stage P (C-layout) -> LDS row-major 16x32 bf16 first, so the reload
    // latency overlaps the O-rescale below
#pragma unroll
    for (int r = 0; r < 8; ++r) {
      const int prow = r + hlf * 8;
      Pw[prow * 32 + lcol]      = f2bf(p0[r]);
      Pw[prow * 32 + 16 + lcol] = f2bf(p1[r]);
    }

#pragma unroll
    for (int t = 0; t < 4; ++t)
#pragma unroll
      for (int r = 0; r < 8; ++r) o[t][r] *= alpha[r];

    asm volatile("s_wait_dscnt 0" ::: "memory");  // same-wave store->load order
    AFrag pa;
    {
      const unsigned short* pp = Pw + lcol * 32 + hlf * 16;
      pa.q[0] = *(const uint4*)(pp);
      pa.q[1] = *(const uint4*)(pp + 8);
    }

    // O += P(16x32) @ V(32x64)
#pragma unroll
    for (int t = 0; t < 4; ++t) o[t] = wmma_bf16(pa.v, vf[t].v, o[t]);
  };

  // All blocks except the last satisfy j0+31 <= q0 -> mask-free.
  const int jlast = ((q0 + 15) >> 5) << 5;
#pragma clang loop unroll(disable)
  for (int j0 = 0; j0 < jlast; j0 += 32) body(FalseT{}, j0);
  body(TrueT{}, jlast);  // single diagonal block

  // normalize and store merged-head bf16
  float inv[8];
#pragma unroll
  for (int r = 0; r < 8; ++r) inv[r] = 1.0f / lr[r];
#pragma unroll
  for (int t = 0; t < 4; ++t) {
#pragma unroll
    for (int r = 0; r < 8; ++r) {
      const int rowg = q0 + r + hlf * 8;
      const int col = h * HD + t * 16 + lcol;
      Ao[(size_t)(b * S_LEN + rowg) * D_DIM + col] = f2bf(o[t][r] * inv[r]);
    }
  }
}

// ------------------------------------------------------------------ driver
extern "C" void kernel_launch(void* const* d_in, const int* in_sizes, int n_in,
                              void* d_out, int out_size, void* d_ws,
                              size_t ws_size, hipStream_t stream) {
  (void)in_sizes; (void)n_in; (void)out_size; (void)ws_size;
  const float* query = (const float*)d_in[0];
  const float* key   = (const float*)d_in[1];
  const float* value = (const float*)d_in[2];
  // d_in[3] = causal mask: handled analytically, unused
  const float* Wq = (const float*)d_in[4];
  const float* bq = (const float*)d_in[5];
  const float* Wk = (const float*)d_in[6];
  const float* bk = (const float*)d_in[7];
  const float* Wv = (const float*)d_in[8];
  const float* bv = (const float*)d_in[9];
  const float* Wo = (const float*)d_in[10];
  const float* bo = (const float*)d_in[11];

  char* ws = (char*)d_ws;
  const size_t XMB = (size_t)M_ROWS * D_DIM * 2;  // 16 MiB bf16 activation
  const size_t WMB = (size_t)D_DIM * D_DIM * 2;   // 2 MiB bf16 weight
  unsigned short* Xq  = (unsigned short*)(ws);
  unsigned short* Xk  = (unsigned short*)(ws + XMB);
  unsigned short* Xv  = (unsigned short*)(ws + 2 * XMB);
  unsigned short* Wqb = (unsigned short*)(ws + 3 * XMB);
  unsigned short* Wkb = (unsigned short*)(ws + 3 * XMB + WMB);
  unsigned short* Wvb = (unsigned short*)(ws + 3 * XMB + 2 * WMB);
  unsigned short* Wob = (unsigned short*)(ws + 3 * XMB + 3 * WMB);
  unsigned short* Qb  = (unsigned short*)(ws + 3 * XMB + 4 * WMB);
  unsigned short* Kb  = (unsigned short*)(ws + 4 * XMB + 4 * WMB);
  unsigned short* Vt  = (unsigned short*)(ws + 5 * XMB + 4 * WMB);
  unsigned short* Ao  = (unsigned short*)(ws + 6 * XMB + 4 * WMB);

  const int nX8 = (M_ROWS * D_DIM) / 8;  // 1048576
  const int nW8 = (D_DIM * D_DIM) / 8;   // 131072
  dim3 cvb(256), cvg(1024);
  cvt_bf16_kernel<<<cvg, cvb, 0, stream>>>(query, (unsigned int*)Xq, nX8);
  cvt_bf16_kernel<<<cvg, cvb, 0, stream>>>(key,   (unsigned int*)Xk, nX8);
  cvt_bf16_kernel<<<cvg, cvb, 0, stream>>>(value, (unsigned int*)Xv, nX8);
  cvt_bf16_kernel<<<cvg, cvb, 0, stream>>>(Wq, (unsigned int*)Wqb, nW8);
  cvt_bf16_kernel<<<cvg, cvb, 0, stream>>>(Wk, (unsigned int*)Wkb, nW8);
  cvt_bf16_kernel<<<cvg, cvb, 0, stream>>>(Wv, (unsigned int*)Wvb, nW8);
  cvt_bf16_kernel<<<cvg, cvb, 0, stream>>>(Wo, (unsigned int*)Wob, nW8);

  dim3 gb(128), gg(M_ROWS / 128, D_DIM / 64);  // (64, 16)
  gemm_kernel<0><<<gg, gb, 0, stream>>>(Xq, Wqb, bq, (void*)Qb);
  gemm_kernel<0><<<gg, gb, 0, stream>>>(Xk, Wkb, bk, (void*)Kb);
  gemm_kernel<1><<<gg, gb, 0, stream>>>(Xv, Wvb, bv, (void*)Vt);

  attn_kernel<<<dim3(S_LEN / 64, B_NUM * H_NUM), dim3(128), 0, stream>>>(
      Qb, Kb, Vt, Ao);

  gemm_kernel<2><<<gg, gb, 0, stream>>>(Ao, Wob, bo, d_out);
}
